// MambaLayer_new_70231305224424
// MI455X (gfx1250) — compile-verified, matrix operand order checked by reading, not run on hardware
//
#include <hip/hip_runtime.h>
#include <hip/hip_bf16.h>

#define D_MODEL 256
#define D_STATE 16
#define D_CONV 4
#define D_INNER 512
#define DT_RANK 16
#define BATCH 4
#define LSEQ 4096
#define NROW (BATCH * LSEQ) /* 16384 */

typedef __bf16 bf16_t;
typedef __attribute__((ext_vector_type(16))) bf16_t v16bf;
typedef __attribute__((ext_vector_type(8)))  float  v8f;

__device__ __forceinline__ float fast_silu(float x) {
  return x / (1.f + __expf(-x));
}

// XOR butterfly via ds_swizzle (group-of-32 mode: and=0x1f, or=0, xor=MASK).
template <int MASK>
__device__ __forceinline__ float bfly_xor(float v) {
  return __int_as_float(
      __builtin_amdgcn_ds_swizzle(__float_as_int(v), (MASK << 10) | 0x1f));
}

// Load a 16x32 WMMA fragment (A side) from f32 row-major data, cvt -> bf16.
// CDNA5 ISA 7.12.2 (16-bit A 16x32):
//   lanes 0-15  (h=0): slots 0..7 hold K=kb+0..7,   slots 8..15 hold K=kb+16..23
//   lanes 16-31 (h=1): slots 0..7 hold K=kb+8..15,  slots 8..15 hold K=kb+24..31
// B is symmetric with r = N (row of W, since B = W^T is K-contiguous in W rows).
__device__ __forceinline__ v16bf load_frag(const float* base, int ld, int kbase) {
  const int lane = threadIdx.x & 31;
  const int r = lane & 15;
  const int h = lane >> 4;
  const float* p = base + (size_t)r * ld + kbase + h * 8;
  v16bf f;
#pragma unroll
  for (int i = 0; i < 8; i++) f[i] = (bf16_t)p[i];
#pragma unroll
  for (int i = 0; i < 8; i++) f[8 + i] = (bf16_t)p[16 + i];
  return f;
}

// Load a pre-packed fragment-major bf16 B fragment: lane's 16 values are
// contiguous (32 B, aligned) -> two b128 loads, no conversion, no gather.
__device__ __forceinline__ v16bf load_frag_pk(const bf16_t* pk, int frag) {
  const int lane = threadIdx.x & 31;
  return *(const v16bf*)(pk + ((size_t)frag * 32 + lane) * 16);
}

__device__ __forceinline__ v8f wmma_bf16(v16bf a, v16bf b, v8f c) {
  return __builtin_amdgcn_wmma_f32_16x16x32_bf16(false, a, false, b, (short)0, c,
                                                 false, false);
}

// f32 16x16 C/D tile: lane n (0-15) holds rows 0..7 of column n; lane n+16 rows 8..15.
__device__ __forceinline__ void store_tile(float* out, int ldc, v8f acc) {
  const int lane = threadIdx.x & 31;
  const int n = lane & 15;
  const int hi = lane >> 4;
#pragma unroll
  for (int r = 0; r < 8; r++) out[(size_t)(r + 8 * hi) * ldc + n] = acc[r];
}

__device__ __forceinline__ void store_tile_silu(float* out, int ldc, v8f acc) {
  const int lane = threadIdx.x & 31;
  const int n = lane & 15;
  const int hi = lane >> 4;
#pragma unroll
  for (int r = 0; r < 8; r++)
    out[(size_t)(r + 8 * hi) * ldc + n] = fast_silu(acc[r]);
}

// ---------------------------------------------------------------------------
// K0: repack weight matrix W [N,K] (f32 row-major) into fragment-major bf16.
//     One thread per (frag, lane); frag = nt*KT + kt.  Output: 16 contiguous
//     bf16 per lane in WMMA slot order.
// ---------------------------------------------------------------------------
__global__ void __launch_bounds__(256)
k_pack_w(const float* __restrict__ W, bf16_t* __restrict__ pk, int NT, int KT,
         int K) {
  const int t = blockIdx.x * 256 + threadIdx.x;
  if (t >= NT * KT * 32) return;
  const int lane = t & 31;
  const int frag = t >> 5;
  const int kt = frag % KT;
  const int nt = frag / KT;
  const int r = lane & 15;
  const int h = lane >> 4;
  const float* p = W + (size_t)(nt * 16 + r) * K + kt * 32 + h * 8;
  bf16_t* o = pk + (size_t)t * 16;
#pragma unroll
  for (int i = 0; i < 8; i++) o[i] = (bf16_t)p[i];
#pragma unroll
  for (int i = 0; i < 8; i++) o[8 + i] = (bf16_t)p[16 + i];
}

// ---------------------------------------------------------------------------
// K1: LayerNorm(256) fused with xz = xn @ W_in^T.  One block = 16 rows.
//     cols [0,512) -> xi (raw), cols [512,1024) -> zs = silu(z).
// ---------------------------------------------------------------------------
__global__ void __launch_bounds__(128)
k_ln_gemm_in(const float* __restrict__ x, const float* __restrict__ g,
             const float* __restrict__ bta, const bf16_t* __restrict__ pk_in,
             float* __restrict__ xi, float* __restrict__ zs) {
  __shared__ float xn[16][260];  // padded row stride
  const int m0 = blockIdx.x * 16;
  const int tid = threadIdx.x;

  // ---- LayerNorm: 8 lanes per row, 32 elements per lane ----
  const int r = tid >> 3;   // row 0..15
  const int sub = tid & 7;  // 0..7
  const float* xr = x + (size_t)(m0 + r) * D_MODEL + sub * 32;
  float v[32];
#pragma unroll
  for (int j = 0; j < 8; j++) {
    float4 q = ((const float4*)xr)[j];
    v[4 * j + 0] = q.x; v[4 * j + 1] = q.y;
    v[4 * j + 2] = q.z; v[4 * j + 3] = q.w;
  }
  float s = 0.f, ss = 0.f;
#pragma unroll
  for (int j = 0; j < 32; j++) { s += v[j]; ss += v[j] * v[j]; }
  s  += bfly_xor<1>(s);  s  += bfly_xor<2>(s);  s  += bfly_xor<4>(s);
  ss += bfly_xor<1>(ss); ss += bfly_xor<2>(ss); ss += bfly_xor<4>(ss);
  const float mu = s * (1.f / D_MODEL);
  const float var = ss * (1.f / D_MODEL) - mu * mu;
  const float rs = rsqrtf(var + 1e-5f);
#pragma unroll
  for (int j = 0; j < 32; j++) {
    const int c = sub * 32 + j;
    xn[r][c] = (v[j] - mu) * rs * g[c] + bta[c];
  }
  __syncthreads();

  // ---- WMMA GEMM: [16x256] x [256x1024]; 4 waves x 16 n-tiles ----
  const int wave = tid >> 5;
  v16bf afr[8];
#pragma unroll
  for (int kt = 0; kt < 8; kt++) afr[kt] = load_frag(&xn[0][0], 260, kt * 32);

  for (int nt = wave; nt < 64; nt += 4) {
    const int e0 = nt * 16;
    v8f acc = {};
#pragma unroll
    for (int kt = 0; kt < 8; kt++) {
      v16bf bfr = load_frag_pk(pk_in, nt * 8 + kt);
      acc = wmma_bf16(afr[kt], bfr, acc);
    }
    if (e0 < D_INNER)
      store_tile(xi + (size_t)m0 * D_INNER + e0, D_INNER, acc);
    else
      store_tile_silu(zs + (size_t)m0 * D_INNER + (e0 - D_INNER), D_INNER, acc);
  }
}

// ---------------------------------------------------------------------------
// K2: causal depthwise conv (k=4) along L per batch + SiLU. float4 over d.
// ---------------------------------------------------------------------------
__global__ void __launch_bounds__(256)
k_conv_silu(const float* __restrict__ xi, const float* __restrict__ cw,
            const float* __restrict__ cb, float* __restrict__ xc) {
  const int t = blockIdx.x * 256 + threadIdx.x;
  const int d = (t & 127) * 4;
  const int l = (t >> 7) & (LSEQ - 1);
  const int b = t >> 19;  // t / (128*4096)
  float w[4][4];
#pragma unroll
  for (int j = 0; j < 4; j++) {
    float4 q = *((const float4*)(cw + (size_t)(d + j) * 4));
    w[j][0] = q.x; w[j][1] = q.y; w[j][2] = q.z; w[j][3] = q.w;
  }
  float4 cbq = *((const float4*)(cb + d));
  float a0 = cbq.x, a1 = cbq.y, a2 = cbq.z, a3 = cbq.w;
  const size_t base_b = (size_t)b * LSEQ * D_INNER + d;
#pragma unroll
  for (int k = 0; k < 4; k++) {
    const int ll = l - (D_CONV - 1) + k;
    if (ll >= 0) {
      float4 q = *((const float4*)(xi + base_b + (size_t)ll * D_INNER));
      a0 += q.x * w[0][k]; a1 += q.y * w[1][k];
      a2 += q.z * w[2][k]; a3 += q.w * w[3][k];
    }
  }
  float4 o;
  o.x = fast_silu(a0); o.y = fast_silu(a1);
  o.z = fast_silu(a2); o.w = fast_silu(a3);
  *((float4*)(xc + base_b + (size_t)l * D_INNER)) = o;
}

// ---------------------------------------------------------------------------
// K3: x_dbl = xc @ W_xp^T  -> [16384, 48].  Wave = one 16-row tile, 3 n-tiles.
// ---------------------------------------------------------------------------
__global__ void __launch_bounds__(128)
k_gemm_xp(const float* __restrict__ xc, const bf16_t* __restrict__ pk_xp,
          float* __restrict__ xdbl) {
  const int wave = threadIdx.x >> 5;
  const int m0 = blockIdx.x * 64 + wave * 16;
  v8f acc0 = {}, acc1 = {}, acc2 = {};
#pragma unroll
  for (int kt = 0; kt < 16; kt++) {
    v16bf a = load_frag(xc + (size_t)m0 * D_INNER, D_INNER, kt * 32);
    v16bf b0 = load_frag_pk(pk_xp, 0 * 16 + kt);
    v16bf b1 = load_frag_pk(pk_xp, 1 * 16 + kt);
    v16bf b2 = load_frag_pk(pk_xp, 2 * 16 + kt);
    acc0 = wmma_bf16(a, b0, acc0);
    acc1 = wmma_bf16(a, b1, acc1);
    acc2 = wmma_bf16(a, b2, acc2);
  }
  store_tile(xdbl + (size_t)m0 * 48 + 0, 48, acc0);
  store_tile(xdbl + (size_t)m0 * 48 + 16, 48, acc1);
  store_tile(xdbl + (size_t)m0 * 48 + 32, 48, acc2);
}

// ---------------------------------------------------------------------------
// K4: selective scan.  lane = state s (16 lanes per (b,d) chain).
//     dt-rank dot and Σ_s h·C both via ds_swizzle butterflies.
//     Fuses: softplus(dt), recurrence, y, D-skip, z-gate.
// ---------------------------------------------------------------------------
__global__ void __launch_bounds__(256)
k_scan(const float* __restrict__ xdbl, const float* __restrict__ xc,
       const float* __restrict__ zs, const float* __restrict__ W_dt,
       const float* __restrict__ b_dt, const float* __restrict__ A_log,
       const float* __restrict__ Dp, float* __restrict__ yg) {
  const int t = blockIdx.x * 256 + threadIdx.x;
  const int s = t & 15;
  const int half = t >> 4;  // 0..2047
  const int d = half & (D_INNER - 1);
  const int b = half >> 9;

  const float A_s = -__expf(A_log[(size_t)d * D_STATE + s]);
  const float wdt = W_dt[(size_t)d * DT_RANK + s];
  const float bdt = b_dt[d];
  const float Dd = Dp[d];
  const float* xrow = xdbl + (size_t)b * LSEQ * 48;
  const float* xcb = xc + (size_t)b * LSEQ * D_INNER + d;
  const float* zsb = zs + (size_t)b * LSEQ * D_INNER + d;
  float* ygb = yg + (size_t)b * LSEQ * D_INNER + d;

  float h = 0.f;
  for (int l = 0; l < LSEQ; l++) {
    const float* row = xrow + (size_t)l * 48;
    if (l + 8 < LSEQ) {
      __builtin_prefetch(xcb + (size_t)(l + 8) * D_INNER, 0, 1);
      __builtin_prefetch(row + 8 * 48, 0, 1);
    }
    // dt = softplus( dot(x_dbl[l, 0:16], W_dt[d,:]) + b_dt[d] )
    float p = row[s] * wdt;
    p += bfly_xor<1>(p); p += bfly_xor<2>(p);
    p += bfly_xor<4>(p); p += bfly_xor<8>(p);
    const float dtp = p + bdt;
    const float dtv = (dtp > 20.f) ? dtp : __logf(1.f + __expf(dtp));

    const float Bv = row[DT_RANK + s];
    const float Cv = row[DT_RANK + D_STATE + s];
    const float xv = xcb[(size_t)l * D_INNER];

    const float dA = __expf(dtv * A_s);
    h = fmaf(dA, h, dtv * xv * Bv);

    float c = h * Cv;
    c += bfly_xor<1>(c); c += bfly_xor<2>(c);
    c += bfly_xor<4>(c); c += bfly_xor<8>(c);
    if (s == 0) {
      const float y = c + Dd * xv;
      ygb[(size_t)l * D_INNER] = y * zsb[(size_t)l * D_INNER];
    }
  }
}

// ---------------------------------------------------------------------------
// K5: out = yg @ W_out^T -> [16384, 256] written straight to d_out
//     (raw reshape => flat layout matches [B,C,H,W] of the reference).
// ---------------------------------------------------------------------------
__global__ void __launch_bounds__(128)
k_gemm_out(const float* __restrict__ yg, const bf16_t* __restrict__ pk_out,
           float* __restrict__ out) {
  const int wave = threadIdx.x >> 5;
  const int m0 = blockIdx.x * 16;
  v8f acc[4] = {};
#pragma unroll
  for (int kt = 0; kt < 16; kt++) {
    v16bf a = load_frag(yg + (size_t)m0 * D_INNER, D_INNER, kt * 32);
#pragma unroll
    for (int i = 0; i < 4; i++) {
      const int nt = wave * 4 + i;
      v16bf bfr = load_frag_pk(pk_out, nt * 16 + kt);
      acc[i] = wmma_bf16(a, bfr, acc[i]);
    }
  }
#pragma unroll
  for (int i = 0; i < 4; i++) {
    const int nt = wave * 4 + i;
    store_tile(out + (size_t)m0 * D_MODEL + nt * 16, D_MODEL, acc[i]);
  }
}

// ---------------------------------------------------------------------------
extern "C" void kernel_launch(void* const* d_in, const int* in_sizes, int n_in,
                              void* d_out, int out_size, void* d_ws,
                              size_t ws_size, hipStream_t stream) {
  const float* x     = (const float*)d_in[0];
  const float* ln_g  = (const float*)d_in[1];
  const float* ln_b  = (const float*)d_in[2];
  const float* W_in  = (const float*)d_in[3];
  const float* cw    = (const float*)d_in[4];
  const float* cb    = (const float*)d_in[5];
  const float* W_xp  = (const float*)d_in[6];
  const float* W_dt  = (const float*)d_in[7];
  const float* bdt   = (const float*)d_in[8];
  const float* A_log = (const float*)d_in[9];
  const float* Dp    = (const float*)d_in[10];
  const float* W_out = (const float*)d_in[11];

  float* ws = (float*)d_ws;
  float* xi   = ws;                                  // [16384,512]
  float* zs   = ws + (size_t)1 * NROW * D_INNER;     // [16384,512] silu(z)
  float* xc   = ws + (size_t)2 * NROW * D_INNER;     // [16384,512]
  float* xdbl = ws + (size_t)3 * NROW * D_INNER;     // [16384,48]
  float* yg   = xi;  // xi dead after conv -> reuse for gated y

  // packed bf16 weights (fragment-major), after xdbl (offsets in floats)
  size_t off = (size_t)3 * NROW * D_INNER + (size_t)NROW * 48;
  bf16_t* pk_in  = (bf16_t*)(ws + off);              // 1024x256  -> 512 KB
  off += (size_t)(2 * D_INNER) * D_MODEL / 2;
  bf16_t* pk_xp  = (bf16_t*)(ws + off);              // 48x512    -> 48 KB
  off += (size_t)(DT_RANK + 2 * D_STATE) * D_INNER / 2;
  bf16_t* pk_out = (bf16_t*)(ws + off);              // 256x512   -> 256 KB

  // --- one-shot weight repacks (tiny) ---
  k_pack_w<<<(64 * 8 * 32 + 255) / 256, 256, 0, stream>>>(W_in, pk_in, 64, 8,
                                                          D_MODEL);
  k_pack_w<<<(3 * 16 * 32 + 255) / 256, 256, 0, stream>>>(W_xp, pk_xp, 3, 16,
                                                          D_INNER);
  k_pack_w<<<(16 * 16 * 32 + 255) / 256, 256, 0, stream>>>(W_out, pk_out, 16,
                                                           16, D_INNER);

  k_ln_gemm_in<<<NROW / 16, 128, 0, stream>>>(x, ln_g, ln_b, pk_in, xi, zs);
  k_conv_silu<<<(NROW * (D_INNER / 4)) / 256, 256, 0, stream>>>(xi, cw, cb, xc);
  k_gemm_xp<<<NROW / 64, 128, 0, stream>>>(xc, pk_xp, xdbl);
  k_scan<<<(BATCH * D_INNER * D_STATE) / 256, 256, 0, stream>>>(
      xdbl, xc, zs, W_dt, bdt, A_log, Dp, yg);
  k_gemm_out<<<NROW / 16, 128, 0, stream>>>(yg, pk_out, (float*)d_out);
}
